// ResidualVectorQuantizer_55327768708293
// MI455X (gfx1250) — matrix operand order repaired and checked by reading, not compiled, not run on hardware
//
#include <hip/hip_runtime.h>

#define BDIM 131072
#define EDIM 256
#define NQL  4
#define NEC  256

typedef __attribute__((ext_vector_type(16))) __bf16 v16bf;
typedef __attribute__((ext_vector_type(8)))  float  v8f;

// ---------------------------------------------------------------------------
// Prep: zero loss accumulator, convert codebooks to bf16, compute ||cb||^2.
// ---------------------------------------------------------------------------
__global__ void rvq_prep(const float* __restrict__ srcCb,
                         const float* __restrict__ recCb,
                         float* __restrict__ lossAcc,
                         float* __restrict__ cc,
                         __bf16* __restrict__ cbB) {
  const long tid    = (long)blockIdx.x * blockDim.x + threadIdx.x;
  const long nthr   = (long)gridDim.x * blockDim.x;
  if (tid == 0) *lossAcc = 0.0f;

  const long total = 2L * NQL * NEC * EDIM;   // 524288 bf16 elements
  for (long i = tid; i < total; i += nthr) {
    long br = i / (NQL * NEC * EDIM);
    long r  = i - br * (NQL * NEC * EDIM);
    float v = br ? recCb[r] : srcCb[r];
    cbB[i] = (__bf16)v;
  }

  const int rows = 2 * NQL * NEC;             // 2048 rows of ||cb||^2
  for (int ri = (int)tid; ri < rows; ri += (int)nthr) {
    int br = ri / (NQL * NEC);
    int rr = ri - br * (NQL * NEC);
    const float* src = (br ? recCb : srcCb) + (long)rr * EDIM;
    float s = 0.0f;
    for (int j = 0; j < EDIM; j++) { float v = src[j]; s += v * v; }
    cc[ri] = s;
  }
}

// ---------------------------------------------------------------------------
// Main: one wave32 per 16-row tile per branch; bf16 WMMA distance GEMM,
// shuffle argmin, fp32 residual update in LDS, loss accumulation.
// ---------------------------------------------------------------------------
__global__ __launch_bounds__(128) void rvq_main(
    const float* __restrict__ x,
    const float* __restrict__ srcCb,
    const float* __restrict__ recCb,
    const float* __restrict__ cc,
    const __bf16* __restrict__ cbB,
    float* __restrict__ out,
    float* __restrict__ lossAcc) {

  __shared__ float sres[4][16][EDIM];         // 64 KB: residual tiles, 1/wave

  const int  lane    = threadIdx.x & 31;
  const int  w       = threadIdx.x >> 5;
  const long gw      = (long)blockIdx.x * 4 + w;
  const int  branch  = (int)(gw & 1);
  const long tile    = gw >> 1;
  const long rowBase = tile * 16;

  const float*  xpart  = x + rowBase * (2 * EDIM) + (long)branch * EDIM;
  const float*  cbF    = branch ? recCb : srcCb;
  const __bf16* cbBb   = cbB + (long)branch * NQL * NEC * EDIM;
  const float*  ccb    = cc + branch * NQL * NEC;
  float* outQ   = out + (long)branch * BDIM * EDIM + rowBase * EDIM;
  float* outIdx = out + 2L * BDIM * EDIM + 1
                      + (long)branch * BDIM * NQL + rowBase * NQL;

  // ---- load residual tile (16 rows x 256 cols) into LDS, b128 per lane ----
  {
    const int c0 = lane * 8;
    for (int m = 0; m < 16; m++) {
      const float4* s4 = (const float4*)(xpart + (long)m * (2 * EDIM) + c0);
      float4 a0 = s4[0], a1 = s4[1];
      float4* d4 = (float4*)&sres[w][m][c0];
      d4[0] = a0; d4[1] = a1;
    }
  }

  const int m_a  = lane & 15;        // A-fragment row (ISA 16-bit A layout)
  const int koff = (lane >> 4) * 16; // K sub-offset per lane half
  const int kbB  = (lane >> 4) * 16; // B-fragment K run per lane half
  float lsum = 0.0f;

  for (int lvl = 0; lvl < NQL; lvl++) {
    // ---- build 8 bf16 A fragments (16x32 each) from LDS residual ----
    v16bf afr[8];
#pragma unroll
    for (int kb = 0; kb < 8; kb++) {
      const float* rrow = &sres[w][m_a][kb * 32];
      v16bf a;
#pragma unroll
      for (int j = 0; j < 8; j++) {
        a[j]     = (__bf16)rrow[koff + j];
        a[8 + j] = (__bf16)rrow[8 + koff + j];
      }
      afr[kb] = a;
    }

    float best[8]; int bidx[8];
#pragma unroll
    for (int g = 0; g < 8; g++) { best[g] = 3.4e38f; bidx[g] = 0; }

    // ---- 16 N-tiles x 8 K-chunks of v_wmma_f32_16x16x32_bf16 ----
    for (int nt = 0; nt < 16; nt++) {
      const int n = nt * 16 + (lane & 15);
      v8f acc = {0.f, 0.f, 0.f, 0.f, 0.f, 0.f, 0.f, 0.f};
#pragma unroll
      for (int kb = 0; kb < 8; kb++) {
        const v16bf b = *(const v16bf*)(cbBb + ((long)lvl * NEC + n) * EDIM
                                        + kb * 32 + kbB);
        acc = __builtin_amdgcn_wmma_f32_16x16x32_bf16(
            false, afr[kb], false, b, (short)0, acc, false, false);
      }
      const float ccv = ccb[lvl * NEC + n];   // ||cb_n||^2 (row term dropped)
#pragma unroll
      for (int g = 0; g < 8; g++) {
        float s = __builtin_fmaf(-2.0f, acc[g], ccv);
        if (s < best[g]) { best[g] = s; bidx[g] = n; }
      }
    }

    // ---- argmin across the 16 lanes sharing each row (xor shuffles) ----
#pragma unroll
    for (int g = 0; g < 8; g++) {
#pragma unroll
      for (int off = 1; off < 16; off <<= 1) {
        float ov = __shfl_xor(best[g], off, 32);
        int   oi = __shfl_xor(bidx[g], off, 32);
        if (ov < best[g] || (ov == best[g] && oi < bidx[g])) {
          best[g] = ov; bidx[g] = oi;
        }
      }
    }

    // ---- gather fp32 codebook rows, update residual, accumulate loss ----
    const int c0 = lane * 8;
    for (int m = 0; m < 16; m++) {
      int im = __shfl(bidx[m & 7], (m >> 3) * 16, 32);
      if (lane == 0) outIdx[m * NQL + lvl] = (float)im;
      const float* cbrow = cbF + ((long)lvl * NEC + im) * EDIM + c0;
      float* rrow = &sres[w][m][c0];
#pragma unroll
      for (int j = 0; j < 8; j++) {
        float nr = rrow[j] - cbrow[j];
        rrow[j] = nr;
        lsum += nr * nr;        // (x_q - r_pre)^2 == r_post^2
      }
    }
  }

  // ---- x_q = x_part - residual_final ----
  {
    const int c0 = lane * 8;
    for (int m = 0; m < 16; m++) {
      const float4* xs = (const float4*)(xpart + (long)m * (2 * EDIM) + c0);
      const float4* rs = (const float4*)&sres[w][m][c0];
      float4* od = (float4*)(outQ + (long)m * EDIM + c0);
      float4 x0 = xs[0], x1 = xs[1], r0 = rs[0], r1 = rs[1];
      float4 o0 = {x0.x - r0.x, x0.y - r0.y, x0.z - r0.z, x0.w - r0.w};
      float4 o1 = {x1.x - r1.x, x1.y - r1.y, x1.z - r1.z, x1.w - r1.w};
      od[0] = o0; od[1] = o1;
    }
  }

  // ---- loss wave-reduce + atomic accumulate ----
#pragma unroll
  for (int off = 1; off < 32; off <<= 1) lsum += __shfl_xor(lsum, off, 32);
  if (lane == 0) atomicAdd(lossAcc, lsum);
}

// ---------------------------------------------------------------------------
// Tail: write mean loss = 1.25/(8*B*E) * sum(residual^2 over all levels).
// ---------------------------------------------------------------------------
__global__ void rvq_final(const float* __restrict__ lossAcc,
                          float* __restrict__ out) {
  if (threadIdx.x == 0 && blockIdx.x == 0) {
    out[2L * BDIM * EDIM] =
        (*lossAcc) * (1.25f / (8.0f * (float)BDIM * (float)EDIM));
  }
}

// ---------------------------------------------------------------------------
extern "C" void kernel_launch(void* const* d_in, const int* in_sizes, int n_in,
                              void* d_out, int out_size, void* d_ws, size_t ws_size,
                              hipStream_t stream) {
  (void)in_sizes; (void)n_in; (void)out_size; (void)ws_size;
  const float* x     = (const float*)d_in[0];
  const float* srcCb = (const float*)d_in[1];
  const float* recCb = (const float*)d_in[2];
  float* out = (float*)d_out;

  char*   wsb     = (char*)d_ws;
  float*  lossAcc = (float*)wsb;                               // 64 B slot
  float*  cc      = (float*)(wsb + 64);                        // 2*4*256 f32
  __bf16* cbB     = (__bf16*)(wsb + 64 + 2 * NQL * NEC * sizeof(float));

  rvq_prep<<<256, 256, 0, stream>>>(srcCb, recCb, lossAcc, cc, cbB);

  // waves = (B/16 rows-tiles) * 2 branches = 16384; 4 waves (128 thr) / block
  const int nBlocks = (BDIM / 16) * 2 / 4;                     // 4096
  rvq_main<<<nBlocks, 128, 0, stream>>>(x, srcCb, recCb, cc, cbB, out, lossAcc);

  rvq_final<<<1, 1, 0, stream>>>(lossAcc, out);
}
